// SlotAttentionFrozen_16286515986741
// MI455X (gfx1250) — matrix operand order repaired and checked by reading, not compiled, not run on hardware
//
#include <hip/hip_runtime.h>

typedef __bf16 bf16_t;
typedef __attribute__((ext_vector_type(16))) __bf16 bf16x16;
typedef __attribute__((ext_vector_type(8)))  float  floatx8;

#define TPB 256

constexpr int N_DET = 16384;
constexpr int DIN_C = 256;
constexpr int DDIM  = 1024;
constexpr int KSLOT = 64;
constexpr int HDIM  = 2048;
constexpr int NITER = 3;
constexpr float EPS_C = 1e-8f;
constexpr float THRESH_C = 0.3f;

// ---- CDNA5 async global->LDS path (guarded; falls back to reg staging) ----
#if defined(__gfx1250__) && __has_builtin(__builtin_amdgcn_global_load_async_to_lds_b128)
#define USE_ASYNC_LDS 1
#else
#define USE_ASYNC_LDS 0
#endif

// Builtin wants: (int4 addrspace(1)* src, int4 addrspace(3)* dst, imm off, imm cpol)
typedef int v4i __attribute__((vector_size(16)));
typedef __attribute__((address_space(1))) v4i* gv4i_p;
typedef __attribute__((address_space(3))) v4i* lv4i_p;

__device__ __forceinline__ gv4i_p as_global_v4(const void* p) {
    return (gv4i_p)(unsigned long long)p;
}
__device__ __forceinline__ lv4i_p as_lds_v4(const void* p) {
    // flat->LDS: low 32 bits of a flat shared address are the LDS byte offset
    return (lv4i_p)(unsigned)(unsigned long long)p;
}

template<int NCNT>
__device__ __forceinline__ void wait_asynccnt_imm() {
#if __has_builtin(__builtin_amdgcn_s_wait_asynccnt)
    __builtin_amdgcn_s_wait_asynccnt(NCNT);
#else
    asm volatile("s_wait_asynccnt %0" :: "n"(NCNT) : "memory");
#endif
}

// ---------------------------------------------------------------------------
// bf16 WMMA GEMM: C[M x N] = A[M x K] * B (+bias)(+resid)(relu?)
//   A: bf16 row-major, lda elements per row
//   B: TRANSB=1 -> bf16 [N x K] row-major; element(k,n) = B[n*ldb + k]
//      TRANSB=0 -> bf16 [K x N] row-major (ldb = row stride)
// Block = 256 threads = 8 waves (2 x 4). Tile BM x 128, K-slab 32.
// Double-buffered LDS; async global->LDS loads when available.
// ---------------------------------------------------------------------------
template<int BM, bool TRANSB, bool RELU, bool OUTBF16>
__global__ __launch_bounds__(256)
void gemm_bf16_wmma(const bf16_t* __restrict__ A, const bf16_t* __restrict__ B,
                    const float* __restrict__ bias, const float* __restrict__ resid,
                    float* __restrict__ Cf, bf16_t* __restrict__ Cb,
                    int M, int N, int Kd, int lda, int ldb)
{
    constexpr int BN = 128;
    constexpr int BK = 32;
    constexpr int LP = BK + 8;                 // padded LDS row: 80B (16B-aligned rows)
    __shared__ alignas(16) bf16_t lsA[2][BM * LP];
    __shared__ alignas(16) bf16_t lsB[2][BN * LP];

    const int tid  = threadIdx.x;
    const int lane = tid & 31;
    const int wave = tid >> 5;
    const int wm   = wave >> 2;                // 0..1
    const int wn   = wave & 3;                 // 0..3
    constexpr int WM = BM / 2;
    constexpr int WN = BN / 4;                 // 32
    constexpr int TM = WM / 16;                // 4 (BM=128) or 2 (BM=64)
    constexpr int TN = WN / 16;                // 2
    const int m0 = blockIdx.y * BM;
    const int n0 = blockIdx.x * BN;
    const int half = lane >> 4;
    const int lr   = lane & 15;

    // ---- K-invariant staging geometry ----
    constexpr int ACH = BM * BK / 8 / TPB;     // A 8-elem chunks per thread (1|2)
    constexpr int BCH = BN * BK / 8 / TPB;     // 2
    constexpr int PEND = ACH + (TRANSB ? BCH : 0);  // async ops per wave per stage

    const bf16_t* aSrc[ACH]; int aOff[ACH];
    #pragma unroll
    for (int i = 0; i < ACH; ++i) {
        int c = tid + i * TPB;
        int row = c >> 2, kc = (c & 3) << 3;
        aSrc[i] = A + (size_t)(m0 + row) * lda + kc;
        aOff[i] = row * LP + kc;
    }
    const bf16_t* bSrc[BCH]; int bOff[BCH]; int bKk[BCH];
    #pragma unroll
    for (int i = 0; i < BCH; ++i) {
        int c = tid + i * TPB;
        if (TRANSB) {                          // contiguous K slab per output col
            int nr = c >> 2, kc = (c & 3) << 3;
            bSrc[i] = B + (size_t)(n0 + nr) * ldb + kc;
            bOff[i] = nr * LP + kc;
        } else {                               // [K x N] source: scatter-transpose
            int kk = c >> 4, nc = (c & 15) << 3;
            bSrc[i] = B + (size_t)kk * ldb + n0 + nc;
            bOff[i] = nc;                      // n-base within tile
            bKk[i]  = kk;
        }
    }

    auto stage = [&](int kt, int sel) {
        const size_t ko = (size_t)kt * BK;
        #pragma unroll
        for (int i = 0; i < ACH; ++i) {
            const bf16_t* src = aSrc[i] + ko;
            bf16_t* dst = &lsA[sel][aOff[i]];
#if USE_ASYNC_LDS
            __builtin_amdgcn_global_load_async_to_lds_b128(
                as_global_v4(src), as_lds_v4(dst), 0, 0);
#else
            *reinterpret_cast<uint4*>(dst) = *reinterpret_cast<const uint4*>(src);
#endif
        }
        if (TRANSB) {
            #pragma unroll
            for (int i = 0; i < BCH; ++i) {
                const bf16_t* src = bSrc[i] + ko;
                bf16_t* dst = &lsB[sel][bOff[i]];
#if USE_ASYNC_LDS
                __builtin_amdgcn_global_load_async_to_lds_b128(
                    as_global_v4(src), as_lds_v4(dst), 0, 0);
#else
                *reinterpret_cast<uint4*>(dst) = *reinterpret_cast<const uint4*>(src);
#endif
            }
        } else {
            #pragma unroll
            for (int i = 0; i < BCH; ++i) {
                uint4 d = *reinterpret_cast<const uint4*>(bSrc[i] + ko * ldb);
                const bf16_t* e = reinterpret_cast<const bf16_t*>(&d);
                #pragma unroll
                for (int j = 0; j < 8; ++j)
                    lsB[sel][(bOff[i] + j) * LP + bKk[i]] = e[j];
            }
        }
    };

    floatx8 acc[TM][TN];
    #pragma unroll
    for (int i = 0; i < TM; ++i)
        #pragma unroll
        for (int j = 0; j < TN; ++j)
            #pragma unroll
            for (int r = 0; r < 8; ++r) acc[i][j][r] = 0.0f;

    const int KT = Kd / BK;
    stage(0, 0);

    for (int kt = 0; kt < KT; ++kt) {
        const int cur = kt & 1;
        if (kt + 1 < KT) stage(kt + 1, cur ^ 1);
#if USE_ASYNC_LDS
        if (kt + 1 < KT) wait_asynccnt_imm<PEND>();   // tile kt complete, kt+1 in flight
        else             wait_asynccnt_imm<0>();
#endif
        __syncthreads();

        union Frag { bf16x16 v; unsigned u[8]; };
        Frag af[TM], bf[TN];
        #pragma unroll
        for (int t = 0; t < TM; ++t) {
            const bf16_t* rp = &lsA[cur][(wm * WM + t * 16 + lr) * LP];
            #pragma unroll
            for (int v = 0; v < 8; ++v) {
                // A 16x32 bf16 layout: K = (v/4)*16 + half*8 + (v%4)*2
                int kk = ((v >> 2) << 4) + (half << 3) + ((v & 3) << 1);
                af[t].u[v] = *reinterpret_cast<const unsigned*>(rp + kk);
            }
        }
        #pragma unroll
        for (int t = 0; t < TN; ++t) {
            const bf16_t* cp = &lsB[cur][(wn * WN + t * 16 + lr) * LP];
            #pragma unroll
            for (int v = 0; v < 8; ++v) {
                // B 32x16 bf16 layout: K = (v/4)*16 + half*8 + (v%4)*2
                int kk = ((v >> 2) << 4) + (half << 3) + ((v & 3) << 1);
                bf[t].u[v] = *reinterpret_cast<const unsigned*>(cp + kk);
            }
        }
        #pragma unroll
        for (int i = 0; i < TM; ++i)
            #pragma unroll
            for (int j = 0; j < TN; ++j)
                acc[i][j] = __builtin_amdgcn_wmma_f32_16x16x32_bf16(
                    false, af[i].v, false, bf[j].v,
                    (short)0, acc[i][j], false, false);
        __syncthreads();
    }

    // ---- epilogue: C/D layout row = r + 8*half, col = lane&15 ----
    #pragma unroll
    for (int i = 0; i < TM; ++i) {
        int mBase = m0 + wm * WM + i * 16 + (half << 3);
        #pragma unroll
        for (int j = 0; j < TN; ++j) {
            int col = n0 + wn * WN + j * 16 + lr;
            float bv = bias ? bias[col] : 0.0f;
            #pragma unroll
            for (int r = 0; r < 8; ++r) {
                int row = mBase + r;
                float vv = acc[i][j][r] + bv;
                if (resid) vv += resid[(size_t)row * N + col];
                if (RELU)  vv = vv > 0.0f ? vv : 0.0f;
                if (OUTBF16) Cb[(size_t)row * N + col] = (bf16_t)vv;
                else         Cf[(size_t)row * N + col] = vv;
            }
        }
    }
}

// ---------------------------------------------------------------------------
// Elementwise / reduction kernels
// ---------------------------------------------------------------------------
__global__ void cast_f32_to_bf16(const float* __restrict__ s, bf16_t* __restrict__ d, int n)
{
    int i = blockIdx.x * TPB + threadIdx.x;
    if (i < n) d[i] = (bf16_t)s[i];
}

// out[c*rows + r] = (bf16)in[r*cols + c]   (rows, cols multiples of 32)
__global__ __launch_bounds__(256)
void transpose_cast_bf16(const float* __restrict__ in, bf16_t* __restrict__ out,
                         int rows, int cols)
{
    __shared__ bf16_t tile[32][34];
    int c0 = blockIdx.x * 32, r0 = blockIdx.y * 32;
    int tx = threadIdx.x & 31, ty = threadIdx.x >> 5;
    #pragma unroll
    for (int rr = ty; rr < 32; rr += 8)
        tile[rr][tx] = (bf16_t)in[(size_t)(r0 + rr) * cols + c0 + tx];
    __syncthreads();
    #pragma unroll
    for (int cc = ty; cc < 32; cc += 8)
        out[(size_t)(c0 + cc) * rows + r0 + tx] = tile[tx][cc];
}

__global__ __launch_bounds__(256)
void layernorm_to_bf16(const float* __restrict__ X, bf16_t* __restrict__ Y, int cols)
{
    __shared__ float red[TPB];
    const int row = blockIdx.x;
    const float* xr = X + (size_t)row * cols;
    float s = 0.f;
    for (int c = threadIdx.x; c < cols; c += TPB) s += xr[c];
    red[threadIdx.x] = s; __syncthreads();
    for (int o = TPB / 2; o > 0; o >>= 1) {
        if (threadIdx.x < o) red[threadIdx.x] += red[threadIdx.x + o];
        __syncthreads();
    }
    float mu = red[0] / cols;
    __syncthreads();
    float s2 = 0.f;
    for (int c = threadIdx.x; c < cols; c += TPB) { float d = xr[c] - mu; s2 += d * d; }
    red[threadIdx.x] = s2; __syncthreads();
    for (int o = TPB / 2; o > 0; o >>= 1) {
        if (threadIdx.x < o) red[threadIdx.x] += red[threadIdx.x + o];
        __syncthreads();
    }
    float inv = rsqrtf(red[0] / cols + 1e-5f);
    for (int c = threadIdx.x; c < cols; c += TPB)
        Y[(size_t)row * cols + c] = (bf16_t)((xr[c] - mu) * inv);
}

// softmax over slot axis (64 rows) per detection column; scale = D^-0.5
__global__ __launch_bounds__(256)
void softmax_cols(float* __restrict__ dots, int ncols, float scale)
{
    int n = blockIdx.x * TPB + threadIdx.x;
    if (n >= ncols) return;
    float vals[KSLOT];
    float mx = -3.4e38f;
    #pragma unroll 8
    for (int i = 0; i < KSLOT; ++i) {
        float v = dots[(size_t)i * ncols + n] * scale;
        vals[i] = v; mx = v > mx ? v : mx;
    }
    float sum = 0.f;
    #pragma unroll 8
    for (int i = 0; i < KSLOT; ++i) { float e = __expf(vals[i] - mx); vals[i] = e; sum += e; }
    float inv = 1.f / sum;
    #pragma unroll 8
    for (int i = 0; i < KSLOT; ++i) dots[(size_t)i * ncols + n] = vals[i] * inv;
}

__global__ __launch_bounds__(256)
void row_sum(const float* __restrict__ A, float* __restrict__ out, int cols)
{
    __shared__ float red[TPB];
    const int row = blockIdx.x;
    float s = 0.f;
    for (int c = threadIdx.x; c < cols; c += TPB) s += A[(size_t)row * cols + c];
    red[threadIdx.x] = s; __syncthreads();
    for (int o = TPB / 2; o > 0; o >>= 1) {
        if (threadIdx.x < o) red[threadIdx.x] += red[threadIdx.x + o];
        __syncthreads();
    }
    if (threadIdx.x == 0) out[row] = red[0];
}

__global__ void attn_norm_cast(const float* __restrict__ attn, const float* __restrict__ rs,
                               bf16_t* __restrict__ out, int cols, int n)
{
    int i = blockIdx.x * TPB + threadIdx.x;
    if (i >= n) return;
    int row = i / cols;
    out[i] = (bf16_t)(attn[i] / (rs[row] + EPS_C));
}

__global__ void slots_init_k(const float* __restrict__ mu, const float* __restrict__ noise,
                             float* __restrict__ out)
{
    int i = blockIdx.x * TPB + threadIdx.x;
    if (i < KSLOT * DDIM) out[i] = mu[i % DDIM] + noise[i];
}

__global__ void gru_kernel(const float* __restrict__ gi, const float* __restrict__ gh,
                           const float* __restrict__ h, float* __restrict__ out)
{
    int i = blockIdx.x * TPB + threadIdx.x;
    if (i >= KSLOT * DDIM) return;
    int k = i / DDIM, d = i % DDIM;
    size_t b = (size_t)k * 3 * DDIM;
    float ir = gi[b + d],            hr = gh[b + d];
    float iz = gi[b + DDIM + d],     hz = gh[b + DDIM + d];
    float in = gi[b + 2 * DDIM + d], hn = gh[b + 2 * DDIM + d];
    float r = 1.f / (1.f + __expf(-(ir + hr)));
    float z = 1.f / (1.f + __expf(-(iz + hz)));
    float n = tanhf(in + r * hn);
    out[i] = (1.f - z) * n + z * h[i];
}

__global__ __launch_bounds__(256)
void row_norm(const float* __restrict__ A, float* __restrict__ out, int cols)
{
    __shared__ float red[TPB];
    const int row = blockIdx.x;
    float s = 0.f;
    for (int c = threadIdx.x; c < cols; c += TPB) {
        float v = A[(size_t)row * cols + c]; s += v * v;
    }
    red[threadIdx.x] = s; __syncthreads();
    for (int o = TPB / 2; o > 0; o >>= 1) {
        if (threadIdx.x < o) red[threadIdx.x] += red[threadIdx.x + o];
        __syncthreads();
    }
    if (threadIdx.x == 0) out[row] = sqrtf(red[0]);
}

__global__ __launch_bounds__(256)
void sim_kernel(const float* __restrict__ a, const float* __restrict__ b,
                const float* __restrict__ na, const float* __restrict__ nb,
                float* __restrict__ sim)
{
    __shared__ float red[TPB];
    int i = blockIdx.x >> 6, j = blockIdx.x & 63;
    float s = 0.f;
    for (int c = threadIdx.x; c < DDIM; c += TPB)
        s += a[(size_t)i * DDIM + c] * b[(size_t)j * DDIM + c];
    red[threadIdx.x] = s; __syncthreads();
    for (int o = TPB / 2; o > 0; o >>= 1) {
        if (threadIdx.x < o) red[threadIdx.x] += red[threadIdx.x + o];
        __syncthreads();
    }
    if (threadIdx.x == 0)
        sim[i * KSLOT + j] = red[0] / ((na[i] + EPS_C) * (nb[j] + EPS_C));
}

__global__ void greedy_match_k(const float* __restrict__ sim, int* __restrict__ matches)
{
    if (threadIdx.x != 0 || blockIdx.x != 0) return;
    float ms[KSLOT]; int mi[KSLOT];
    for (int i = 0; i < KSLOT; ++i) {
        float best = -3.4e38f; int bj = 0;
        for (int j = 0; j < KSLOT; ++j) {
            float v = sim[i * KSLOT + j];
            if (v > best) { best = v; bj = j; }
        }
        ms[i] = best; mi[i] = bj;
    }
    int order[KSLOT]; bool taken[KSLOT];
    for (int i = 0; i < KSLOT; ++i) taken[i] = false;
    for (int t = 0; t < KSLOT; ++t) {
        float best = -3.4e38f; int bi = -1;
        for (int i = 0; i < KSLOT; ++i)
            if (!taken[i] && ms[i] > best) { best = ms[i]; bi = i; }
        if (bi < 0) { for (int i = 0; i < KSLOT; ++i) if (!taken[i]) { bi = i; break; } }
        taken[bi] = true; order[t] = bi;
    }
    int mt[KSLOT]; bool used[KSLOT];
    for (int i = 0; i < KSLOT; ++i) { mt[i] = -1; used[i] = false; }
    for (int t = 0; t < KSLOT; ++t) {
        int idx = order[t];
        int j = mi[idx];
        bool ok = (!used[j]) && (ms[idx] > THRESH_C);
        if (ok) { mt[idx] = j; used[j] = true; }
    }
    for (int i = 0; i < KSLOT; ++i) matches[i] = mt[i];
}

// ---------------------------------------------------------------------------
// Host orchestration
// ---------------------------------------------------------------------------
static inline int cdiv(int a, int b) { return (a + b - 1) / b; }

extern "C" void kernel_launch(void* const* d_in, const int* in_sizes, int n_in,
                              void* d_out, int out_size, void* d_ws, size_t ws_size,
                              hipStream_t stream)
{
    const float* det_t   = (const float*)d_in[0];
    const float* det_t1  = (const float*)d_in[1];
    const float* W_enc   = (const float*)d_in[2];
    const float* b_enc   = (const float*)d_in[3];
    const float* Wq      = (const float*)d_in[4];
    const float* bq      = (const float*)d_in[5];
    const float* Wk      = (const float*)d_in[6];
    const float* bk      = (const float*)d_in[7];
    const float* Wv      = (const float*)d_in[8];
    const float* bv      = (const float*)d_in[9];
    const float* Wi      = (const float*)d_in[10];
    const float* Wh      = (const float*)d_in[11];
    const float* bi      = (const float*)d_in[12];
    const float* bh      = (const float*)d_in[13];
    const float* W1      = (const float*)d_in[14];
    const float* b1      = (const float*)d_in[15];
    const float* W2      = (const float*)d_in[16];
    const float* b2      = (const float*)d_in[17];
    const float* slots_mu   = (const float*)d_in[18];
    const float* init_noise = (const float*)d_in[19];

    char* p = (char*)d_ws;
    auto alloc = [&](size_t bytes) -> char* {
        char* r = p; p += (bytes + 255) & ~(size_t)255; return r;
    };
    // transposed bf16 weights ([N x K] so all GEMM B-tiles are contiguous)
    bf16_t* wencT = (bf16_t*)alloc((size_t)DIN_C * DDIM * 2);  // [DDIM x DIN]
    bf16_t* wqT   = (bf16_t*)alloc((size_t)DDIM * DDIM * 2);
    bf16_t* wkT   = (bf16_t*)alloc((size_t)DDIM * DDIM * 2);
    bf16_t* wvT   = (bf16_t*)alloc((size_t)DDIM * DDIM * 2);
    bf16_t* wi_bf = (bf16_t*)alloc((size_t)3 * DDIM * DDIM * 2);  // already [3D x D]
    bf16_t* wh_bf = (bf16_t*)alloc((size_t)3 * DDIM * DDIM * 2);
    bf16_t* w1T   = (bf16_t*)alloc((size_t)DDIM * HDIM * 2);   // [HDIM x DDIM]
    bf16_t* w2T   = (bf16_t*)alloc((size_t)HDIM * DDIM * 2);   // [DDIM x HDIM]
    bf16_t* det_bf  = (bf16_t*)alloc((size_t)N_DET * DIN_C * 2);
    // Big region: xenc (f32) lives here first, then reused as k_bf + v_bf
    char*  big   = alloc((size_t)N_DET * DDIM * 4);
    float* xenc  = (float*)big;
    bf16_t* k_bf = (bf16_t*)big;
    bf16_t* v_bf = (bf16_t*)(big + (size_t)N_DET * DDIM * 2);
    bf16_t* x_bf = (bf16_t*)alloc((size_t)N_DET * DDIM * 2);
    float*  dots = (float*)alloc((size_t)KSLOT * N_DET * 4);
    bf16_t* attn_bf = (bf16_t*)alloc((size_t)KSLOT * N_DET * 2);
    float*  rowsum  = (float*)alloc(KSLOT * 4);
    bf16_t* q_bf  = (bf16_t*)alloc((size_t)KSLOT * DDIM * 2);
    bf16_t* ln_bf = (bf16_t*)alloc((size_t)KSLOT * DDIM * 2);
    bf16_t* u_bf  = (bf16_t*)alloc((size_t)KSLOT * DDIM * 2);
    bf16_t* h_bf  = (bf16_t*)alloc((size_t)KSLOT * DDIM * 2);
    bf16_t* h1_bf = (bf16_t*)alloc((size_t)KSLOT * HDIM * 2);
    float*  gi    = (float*)alloc((size_t)KSLOT * 3 * DDIM * 4);
    float*  gh    = (float*)alloc((size_t)KSLOT * 3 * DDIM * 4);
    float*  sA    = (float*)alloc((size_t)KSLOT * DDIM * 4);
    float*  sB    = (float*)alloc((size_t)KSLOT * DDIM * 4);
    float*  prev  = (float*)alloc((size_t)KSLOT * DDIM * 4);
    float*  norms = (float*)alloc(2 * KSLOT * 4);

    auto cast = [&](const float* s, bf16_t* d, int n) {
        cast_f32_to_bf16<<<cdiv(n, TPB), TPB, 0, stream>>>(s, d, n);
    };
    auto tcast = [&](const float* s, bf16_t* d, int rows, int cols) {
        transpose_cast_bf16<<<dim3(cols / 32, rows / 32), TPB, 0, stream>>>(s, d, rows, cols);
    };

    // ---- weights -> bf16 (transposed where the GEMM wants [N x K]) ----
    tcast(W_enc, wencT, DIN_C, DDIM);
    tcast(Wq,   wqT,  DDIM, DDIM);
    tcast(Wk,   wkT,  DDIM, DDIM);
    tcast(Wv,   wvT,  DDIM, DDIM);
    tcast(W1,   w1T,  DDIM, HDIM);
    tcast(W2,   w2T,  HDIM, DDIM);
    cast(Wi, wi_bf, 3 * DDIM * DDIM);
    cast(Wh, wh_bf, 3 * DDIM * DDIM);

    const float scale = 0.03125f;   // 1024^-0.5

    for (int frame = 0; frame < 2; ++frame) {
        const float* dets = (frame == 0) ? det_t : det_t1;
        cast(dets, det_bf, N_DET * DIN_C);

        // x_enc = dets @ W_enc + b_enc
        gemm_bf16_wmma<128, true, false, false><<<dim3(DDIM / 128, N_DET / 128), TPB, 0, stream>>>(
            det_bf, wencT, b_enc, nullptr, xenc, nullptr,
            N_DET, DDIM, DIN_C, DIN_C, DIN_C);
        layernorm_to_bf16<<<N_DET, TPB, 0, stream>>>(xenc, x_bf, DDIM);
        // k = x @ Wk + bk (bf16 out; reuses xenc region)
        gemm_bf16_wmma<128, true, false, true><<<dim3(DDIM / 128, N_DET / 128), TPB, 0, stream>>>(
            x_bf, wkT, bk, nullptr, nullptr, k_bf,
            N_DET, DDIM, DDIM, DDIM, DDIM);
        // v = x @ Wv + bv
        gemm_bf16_wmma<128, true, false, true><<<dim3(DDIM / 128, N_DET / 128), TPB, 0, stream>>>(
            x_bf, wvT, bv, nullptr, nullptr, v_bf,
            N_DET, DDIM, DDIM, DDIM, DDIM);

        if (frame == 0)
            slots_init_k<<<cdiv(KSLOT * DDIM, TPB), TPB, 0, stream>>>(slots_mu, init_noise, sA);

        for (int it = 0; it < NITER; ++it) {
            layernorm_to_bf16<<<KSLOT, TPB, 0, stream>>>(sA, ln_bf, DDIM);
            // q = LN(slots) @ Wq + bq
            gemm_bf16_wmma<64, true, false, true><<<dim3(DDIM / 128, 1), TPB, 0, stream>>>(
                ln_bf, wqT, bq, nullptr, nullptr, q_bf,
                KSLOT, DDIM, DDIM, DDIM, DDIM);
            // dots = q @ k^T
            gemm_bf16_wmma<64, true, false, false><<<dim3(N_DET / 128, 1), TPB, 0, stream>>>(
                q_bf, k_bf, nullptr, nullptr, dots, nullptr,
                KSLOT, N_DET, DDIM, DDIM, DDIM);
            softmax_cols<<<cdiv(N_DET, TPB), TPB, 0, stream>>>(dots, N_DET, scale);
            row_sum<<<KSLOT, TPB, 0, stream>>>(dots, rowsum, N_DET);
            attn_norm_cast<<<cdiv(KSLOT * N_DET, TPB), TPB, 0, stream>>>(
                dots, rowsum, attn_bf, N_DET, KSLOT * N_DET);
            // updates = attn @ v   (the one [K x N]-layout B)
            gemm_bf16_wmma<64, false, false, true><<<dim3(DDIM / 128, 1), TPB, 0, stream>>>(
                attn_bf, v_bf, nullptr, nullptr, nullptr, u_bf,
                KSLOT, DDIM, N_DET, N_DET, DDIM);
            // GRU gates
            cast(sA, h_bf, KSLOT * DDIM);
            gemm_bf16_wmma<64, true, false, false><<<dim3(3 * DDIM / 128, 1), TPB, 0, stream>>>(
                u_bf, wi_bf, bi, nullptr, gi, nullptr,
                KSLOT, 3 * DDIM, DDIM, DDIM, DDIM);
            gemm_bf16_wmma<64, true, false, false><<<dim3(3 * DDIM / 128, 1), TPB, 0, stream>>>(
                h_bf, wh_bf, bh, nullptr, gh, nullptr,
                KSLOT, 3 * DDIM, DDIM, DDIM, DDIM);
            gru_kernel<<<cdiv(KSLOT * DDIM, TPB), TPB, 0, stream>>>(gi, gh, sA, sB);
            // slots = sB + W2 @ relu(W1 @ LN(sB))
            layernorm_to_bf16<<<KSLOT, TPB, 0, stream>>>(sB, ln_bf, DDIM);
            gemm_bf16_wmma<64, true, true, true><<<dim3(HDIM / 128, 1), TPB, 0, stream>>>(
                ln_bf, w1T, b1, nullptr, nullptr, h1_bf,
                KSLOT, HDIM, DDIM, DDIM, DDIM);
            gemm_bf16_wmma<64, true, false, false><<<dim3(DDIM / 128, 1), TPB, 0, stream>>>(
                h1_bf, w2T, b2, sB, sA, nullptr,
                KSLOT, DDIM, HDIM, HDIM, HDIM);
        }

        if (frame == 0)
            (void)hipMemcpyAsync(prev, sA, (size_t)KSLOT * DDIM * 4,
                                 hipMemcpyDeviceToDevice, stream);
    }

    float* simOut = (float*)d_out + KSLOT;
    int*   mOut   = (int*)d_out;
    row_norm<<<KSLOT, TPB, 0, stream>>>(prev, norms, DDIM);
    row_norm<<<KSLOT, TPB, 0, stream>>>(sA, norms + KSLOT, DDIM);
    sim_kernel<<<KSLOT * KSLOT, TPB, 0, stream>>>(prev, sA, norms, norms + KSLOT, simOut);
    greedy_match_k<<<1, 1, 0, stream>>>(simOut, mOut);
}
